// GraphStateEncoder_34488587387632
// MI455X (gfx1250) — matrix operand order repaired and checked by reading, not compiled, not run on hardware
//
#include <hip/hip_runtime.h>
#include <hip/hip_bf16.h>
#include <math.h>

#define N_NODES 20000
#define N_EDGES 320000
#define E_TOT   (N_EDGES + N_NODES)
#define N_GRAPH 16
#define DIN     256
#define HID     64
#define HEADS   4
#define SLOPE   0.2f

typedef __attribute__((ext_vector_type(16))) _Float16 v16h;
typedef __attribute__((ext_vector_type(8)))  float    v8f;
typedef int vec4i __attribute__((ext_vector_type(4)));

// gfx1250 async global->LDS copy path (ASYNCcnt-tracked), with safe fallback.
// Builtin signature (probe-confirmed via diagnostic): 
//   void (vec4i AS1* gsrc, vec4i AS3* ldst, imm int offset, imm int cpol)
#if __has_builtin(__builtin_amdgcn_global_load_async_to_lds_b128) && \
    __has_builtin(__builtin_amdgcn_s_wait_asynccnt)
#define ASYNC_LDS 1
typedef __attribute__((address_space(1))) vec4i* glb_v4i;
typedef __attribute__((address_space(3))) vec4i* lds_v4i;
#else
#define ASYNC_LDS 0
#endif

// float atomic max via sign-split integer atomics (portable)
__device__ __forceinline__ void atomicMaxF(float* addr, float v) {
    if (v >= 0.0f) atomicMax((int*)addr, __float_as_int(v));
    else           atomicMin((unsigned int*)addr, __float_as_uint(v));
}

// ---------------------------------------------------------------------------
// GEMM: H[N, NOUT] = X[N, K] * W[K, NOUT], fp32 in/out, f16 WMMA, f32 accum.
// block = 128 threads (4 waves). Each block: 16 rows x 64 cols; wave w owns the
// 16x16 tile at col = blockIdx.y*64 + w*16. K fully unrolled in steps of 32,
// A-tile double-buffered through LDS (async copy when available).
// Fragment layouts used by this code (wave32):
//   A (16x32 f16): lane L: M=L%16, elem e: K = ((L>=16)?16:0) + e
//   B (32x16 f16): lane L: N=L%16, elem e: K = ((L>=16)?16:0) + e
//   D (16x16 f32): lane L: N=L%16, VGPR v: M = v + ((L>=16)?8:0)
// ---------------------------------------------------------------------------
template <int K, int NOUT>
__global__ __launch_bounds__(128)
void gemm_wmma_f16(const float* __restrict__ X, const float* __restrict__ Wm,
                   float* __restrict__ H)
{
    __shared__ float As[2][16 * 32];
    constexpr int KSTEPS = K / 32;

    const int t       = threadIdx.x;
    const int lane    = t & 31;
    const int wave    = t >> 5;
    const int rowBase = blockIdx.x * 16;
    const int colBase = blockIdx.y * 64 + wave * 16;
    const bool active = (colBase < NOUT);

    const int col    = colBase + (lane & 15);
    const int mRow   = lane & 15;
    const int kHalfA = (lane >> 4) << 3;   // 0 or 8
    const int kHalfB = (lane >> 4) << 4;   // 0 or 16

    // per-thread staging coords: 4 floats (one b128); 128 thr * 16B = 2KB tile
    const int sr = (t * 4) >> 5;
    const int sc = (t * 4) & 31;

    v8f acc = {};

    auto stage = [&](int step, int buf) {
        const float* gsrc = &X[(size_t)(rowBase + sr) * K + step * 32 + sc];
        float* ldst = &As[buf][t * 4];
#if ASYNC_LDS
        __builtin_amdgcn_global_load_async_to_lds_b128(
            (glb_v4i)gsrc, (lds_v4i)ldst, 0, 0);
#else
        *(float4*)ldst = *(const float4*)gsrc;
#endif
    };
    auto wait_stage = []() {
#if ASYNC_LDS
        __builtin_amdgcn_s_wait_asynccnt(0);
#endif
    };

    stage(0, 0);
    wait_stage();
    __syncthreads();

#pragma unroll
    for (int i = 0; i < KSTEPS; ++i) {
        const int cur = i & 1;
        if (i + 1 < KSTEPS) stage(i + 1, cur ^ 1);   // overlap with compute

        if (active) {
            const float* a_base = &As[cur][mRow * 32];
            v16h a, b;
#pragma unroll
            for (int e = 0; e < 16; ++e) {
                int kk = kHalfB + e;
                a[e] = (_Float16)a_base[kk];
            }
#pragma unroll
            for (int e = 0; e < 16; ++e) {
                b[e] = (_Float16)Wm[(size_t)(i * 32 + kHalfB + e) * NOUT + col];
            }
            acc = __builtin_amdgcn_wmma_f32_16x16x32_f16(
                false, a, false, b, (short)0, acc, false, false);
        }

        wait_stage();
        __syncthreads();
    }

    if (active) {
#pragma unroll
        for (int v = 0; v < 8; ++v) {
            int row = rowBase + v + ((lane >> 4) << 3);
            H[(size_t)row * NOUT + col] = acc[v];
        }
    }
}

// es[n,h] = sum_d h[n,h,d]*a_src[h,d] ; ed likewise
__global__ void node_logits(const float* __restrict__ h,
                            const float* __restrict__ a_src,
                            const float* __restrict__ a_dst,
                            float* __restrict__ es, float* __restrict__ ed,
                            int Hh, int Dd)
{
    int i = blockIdx.x * blockDim.x + threadIdx.x;
    if (i >= N_NODES * Hh) return;
    int n = i / Hh, hd = i - n * Hh;
    const float* hp = &h[(size_t)(n * Hh + hd) * Dd];
    const float* ap = &a_src[hd * Dd];
    const float* bp = &a_dst[hd * Dd];
    float s1 = 0.0f, s2 = 0.0f;
    for (int d = 0; d < Dd; ++d) { float v = hp[d]; s1 += v * ap[d]; s2 += v * bp[d]; }
    es[i] = s1;
    ed[i] = s2;
}

__global__ void init_layer(float* m, float* s, float* agg, int nh, int nf)
{
    int i = blockIdx.x * blockDim.x + threadIdx.x;
    if (i < nh) { m[i] = -1e30f; s[i] = 0.0f; }
    if (i < nf) agg[i] = 0.0f;
}

__device__ __forceinline__ void edge_endpoints(const int* ei, int e, int& src, int& dst)
{
    if (e < N_EDGES) { src = ei[e]; dst = ei[N_EDGES + e]; }
    else             { src = dst = e - N_EDGES; }          // self loops
}

__global__ void edge_max(const int* __restrict__ ei,
                         const float* __restrict__ es, const float* __restrict__ ed,
                         float* __restrict__ m, int Hh)
{
    int i = blockIdx.x * blockDim.x + threadIdx.x;
    if (i >= E_TOT * Hh) return;
    int e = i / Hh, hd = i - e * Hh;
    int src, dst; edge_endpoints(ei, e, src, dst);
    float v = es[src * Hh + hd] + ed[dst * Hh + hd];
    v = v > 0.0f ? v : SLOPE * v;
    atomicMaxF(&m[dst * Hh + hd], v);
}

__global__ void edge_expsum(const int* __restrict__ ei,
                            const float* __restrict__ es, const float* __restrict__ ed,
                            const float* __restrict__ m,
                            float* __restrict__ s, float* __restrict__ exbuf, int Hh)
{
    int i = blockIdx.x * blockDim.x + threadIdx.x;
    if (i >= E_TOT * Hh) return;
    int e = i / Hh, hd = i - e * Hh;
    int src, dst; edge_endpoints(ei, e, src, dst);
    float v = es[src * Hh + hd] + ed[dst * Hh + hd];
    v = v > 0.0f ? v : SLOPE * v;
    float mm = m[dst * Hh + hd];
    if (mm < -1e29f) mm = 0.0f;                // isfinite guard
    float ex = expf(v - mm);
    exbuf[i] = ex;
    atomicAdd(&s[dst * Hh + hd], ex);
}

// 4 features per thread: b128 gather from h[src], 4 scatter-adds to agg[dst]
__global__ void edge_aggregate(const int* __restrict__ ei,
                               const float* __restrict__ s,
                               const float* __restrict__ exbuf,
                               const float* __restrict__ h,
                               float* __restrict__ agg, int Hh, int Dd)
{
    const int F = Hh * Dd;
    const int Q = F >> 2;
    long long i = (long long)blockIdx.x * blockDim.x + threadIdx.x;
    if (i >= (long long)E_TOT * Q) return;
    int e  = (int)(i / Q);
    int f  = (int)(i - (long long)e * Q) << 2;
    int hd = f / Dd;                           // 4-elem group never crosses a head
    int src, dst; edge_endpoints(ei, e, src, dst);
    float a = exbuf[e * Hh + hd] / (s[dst * Hh + hd] + 1e-16f);
    const float4 hv = *(const float4*)&h[(size_t)src * F + f];
    float* ap = &agg[(size_t)dst * F + f];
    atomicAdd(ap + 0, hv.x * a);
    atomicAdd(ap + 1, hv.y * a);
    atomicAdd(ap + 2, hv.z * a);
    atomicAdd(ap + 3, hv.w * a);
}

// out = relu(LN(agg + bias)) [+ resid]; one wave32 per node
__global__ __launch_bounds__(256)
void ln_relu(const float* __restrict__ agg, const float* __restrict__ bias,
             const float* __restrict__ g, const float* __restrict__ be,
             const float* __restrict__ resid, float* __restrict__ out, int F)
{
    int node = blockIdx.x * (blockDim.x >> 5) + (threadIdx.x >> 5);
    int lane = threadIdx.x & 31;
    if (node >= N_NODES) return;
    const float* row = &agg[(size_t)node * F];
    float sum = 0.0f, sq = 0.0f;
    for (int f = lane; f < F; f += 32) {
        float v = row[f] + bias[f];
        sum += v; sq += v * v;
    }
#pragma unroll
    for (int off = 16; off > 0; off >>= 1) {
        sum += __shfl_xor(sum, off, 32);
        sq  += __shfl_xor(sq,  off, 32);
    }
    float mu  = sum / (float)F;
    float var = sq / (float)F - mu * mu;
    float inv = rsqrtf(var + 1e-5f);
    for (int f = lane; f < F; f += 32) {
        float v = (row[f] + bias[f] - mu) * inv * g[f] + be[f];
        v = v > 0.0f ? v : 0.0f;
        if (resid) v += resid[(size_t)node * F + f];
        out[(size_t)node * F + f] = v;
    }
}

__global__ void pool_init(float* psum, float* pmax, float* pcnt)
{
    int i = blockIdx.x * blockDim.x + threadIdx.x;
    if (i < N_GRAPH * HID) { psum[i] = 0.0f; pmax[i] = -1e30f; }
    if (i < N_GRAPH) pcnt[i] = 0.0f;
}

__global__ void pool_accum(const float* __restrict__ x, const int* __restrict__ batch,
                           float* psum, float* pmax, float* pcnt)
{
    int i = blockIdx.x * blockDim.x + threadIdx.x;
    if (i >= N_NODES * HID) return;
    int n = i / HID, f = i - n * HID;
    int gidx = batch[n];
    float v = x[i];
    atomicAdd(&psum[gidx * HID + f], v);
    atomicMaxF(&pmax[gidx * HID + f], v);
    if (f == 0) atomicAdd(&pcnt[gidx], 1.0f);
}

__global__ void pool_write(const float* psum, const float* pmax, const float* pcnt,
                           float* out)
{
    int i = blockIdx.x * blockDim.x + threadIdx.x;
    if (i >= N_GRAPH * 2 * HID) return;
    int gidx = i / (2 * HID), f = i - gidx * 2 * HID;
    float v;
    if (f < HID) v = psum[gidx * HID + f] / fmaxf(pcnt[gidx], 1.0f);
    else {
        v = pmax[gidx * HID + (f - HID)];
        if (v < -1e29f) v = 0.0f;
    }
    out[i] = v;
}

// ---------------------------------------------------------------------------
static void run_gat_layer(const float* x, const float* Wm,
                          const float* asrc, const float* adst,
                          const float* bias, const float* gamma, const float* beta,
                          float* h, float* agg, float* es, float* ed,
                          float* m, float* s, float* exbuf,
                          const int* ei, const float* resid, float* xout,
                          int heads, int dd, hipStream_t stream)
{
    const int F = heads * dd;
    // 1. h = x @ W  (compile-time shapes -> fully unrolled WMMA chain)
    dim3 gg(N_NODES / 16, (F + 63) / 64);
    if (F == DIN)
        gemm_wmma_f16<DIN, DIN><<<gg, dim3(128), 0, stream>>>(x, Wm, h);
    else
        gemm_wmma_f16<DIN, HID><<<gg, dim3(128), 0, stream>>>(x, Wm, h);
    // 2. attention logits per node
    int nh = N_NODES * heads;
    node_logits<<<dim3((nh + 255) / 256), dim3(256), 0, stream>>>(
        h, asrc, adst, es, ed, heads, dd);
    // 3. init m/s/agg
    int nf = N_NODES * F;
    init_layer<<<dim3((nf + 255) / 256), dim3(256), 0, stream>>>(m, s, agg, nh, nf);
    // 4. segment max over edges
    int eh = E_TOT * heads;
    edge_max<<<dim3((eh + 255) / 256), dim3(256), 0, stream>>>(ei, es, ed, m, heads);
    // 5. exp + segment sum
    edge_expsum<<<dim3((eh + 255) / 256), dim3(256), 0, stream>>>(
        ei, es, ed, m, s, exbuf, heads);
    // 6. weighted scatter-add (4 features/thread)
    long long eq = (long long)E_TOT * (F >> 2);
    edge_aggregate<<<dim3((unsigned)((eq + 255) / 256)), dim3(256), 0, stream>>>(
        ei, s, exbuf, h, agg, heads, dd);
    // 7. LN + relu (+ residual)
    ln_relu<<<dim3((N_NODES + 7) / 8), dim3(256), 0, stream>>>(
        agg, bias, gamma, beta, resid, xout, F);
}

extern "C" void kernel_launch(void* const* d_in, const int* in_sizes, int n_in,
                              void* d_out, int out_size, void* d_ws, size_t ws_size,
                              hipStream_t stream)
{
    const float* node_x = (const float*)d_in[0];
    const int*   ei     = (const int*)d_in[1];
    const int*   batch  = (const int*)d_in[2];
    const float* W1  = (const float*)d_in[3];
    const float* as1 = (const float*)d_in[4];
    const float* ad1 = (const float*)d_in[5];
    const float* b1  = (const float*)d_in[6];
    const float* g1  = (const float*)d_in[7];
    const float* be1 = (const float*)d_in[8];
    const float* W2  = (const float*)d_in[9];
    const float* as2 = (const float*)d_in[10];
    const float* ad2 = (const float*)d_in[11];
    const float* b2  = (const float*)d_in[12];
    const float* g2  = (const float*)d_in[13];
    const float* be2 = (const float*)d_in[14];
    const float* Wf  = (const float*)d_in[15];
    const float* asf = (const float*)d_in[16];
    const float* adf = (const float*)d_in[17];
    const float* bf  = (const float*)d_in[18];
    const float* gf  = (const float*)d_in[19];
    const float* bef = (const float*)d_in[20];

    float* ws    = (float*)d_ws;
    float* xcur  = ws;                                   // N*256
    float* h     = xcur  + (size_t)N_NODES * DIN;        // N*256
    float* agg   = h     + (size_t)N_NODES * DIN;        // N*256
    float* es    = agg   + (size_t)N_NODES * DIN;        // N*HEADS
    float* ed    = es    + (size_t)N_NODES * HEADS;
    float* mbuf  = ed    + (size_t)N_NODES * HEADS;
    float* sbuf  = mbuf  + (size_t)N_NODES * HEADS;
    float* exbuf = sbuf  + (size_t)N_NODES * HEADS;      // E_TOT*HEADS
    float* psum  = exbuf + (size_t)E_TOT * HEADS;        // NG*HID
    float* pmax  = psum  + (size_t)N_GRAPH * HID;
    float* pcnt  = pmax  + (size_t)N_GRAPH * HID;        // NG

    float* out_x    = (float*)d_out;                     // [N, HID]
    float* out_pool = out_x + (size_t)N_NODES * HID;     // [NG, 2*HID]

    // x = node_x
    (void)hipMemcpyAsync(xcur, node_x, sizeof(float) * (size_t)N_NODES * DIN,
                         hipMemcpyDeviceToDevice, stream);

    // layer 1: x = relu(LN(gat(x))) + x
    run_gat_layer(xcur, W1, as1, ad1, b1, g1, be1,
                  h, agg, es, ed, mbuf, sbuf, exbuf, ei, xcur, xcur,
                  HEADS, HID, stream);
    // layer 2
    run_gat_layer(xcur, W2, as2, ad2, b2, g2, be2,
                  h, agg, es, ed, mbuf, sbuf, exbuf, ei, xcur, xcur,
                  HEADS, HID, stream);
    // final layer: heads=1, no residual, write straight to d_out
    run_gat_layer(xcur, Wf, asf, adf, bf, gf, bef,
                  h, agg, es, ed, mbuf, sbuf, exbuf, ei, nullptr, out_x,
                  1, HID, stream);

    // pooling: mean || max per graph
    pool_init<<<dim3((N_GRAPH * HID + 255) / 256), dim3(256), 0, stream>>>(
        psum, pmax, pcnt);
    pool_accum<<<dim3((N_NODES * HID + 255) / 256), dim3(256), 0, stream>>>(
        out_x, batch, psum, pmax, pcnt);
    pool_write<<<dim3((N_GRAPH * 2 * HID + 255) / 256), dim3(256), 0, stream>>>(
        psum, pmax, pcnt, out_pool);
}
